// PresidentialAttention_15633680957930
// MI455X (gfx1250) — matrix operand order repaired and checked by reading, not compiled, not run on hardware
//
#include <hip/hip_runtime.h>
#include <hip/hip_bf16.h>

// Problem constants (from reference)
#define BATCH   2
#define SEQ     2048
#define HID     1024
#define NHEADS  16
#define HEADD   64
#define MTOT    (BATCH * SEQ)      // 4096 tokens total
#define SCALE   0.125f             // 1/sqrt(64)

typedef __attribute__((ext_vector_type(16))) __bf16         v16bf;
typedef __attribute__((ext_vector_type(8)))  float          v8f;
typedef __attribute__((ext_vector_type(8)))  unsigned short u16x8;
typedef __attribute__((ext_vector_type(16))) unsigned short u16x16;

union BF16x16 { u16x16 u; v16bf v; };

__device__ __forceinline__ unsigned short f2bf(float f) {
  unsigned u = __builtin_bit_cast(unsigned, f);
  u += 0x7FFFu + ((u >> 16) & 1u);            // round-to-nearest-even
  return (unsigned short)(u >> 16);
}

// A-fragment (16-bit 16x32): per-lane K = kh*16 + 0..15 contiguous.
// Caller passes base + kh*16; we load p[0..7] and p[8..15].
__device__ __forceinline__ v16bf ldA(const unsigned short* p) {
  BF16x16 t;
  u16x8 lo = *(const u16x8*)p;
  u16x8 hi = *(const u16x8*)(p + 8);
#pragma unroll
  for (int i = 0; i < 8; ++i) { t.u[i] = lo[i]; t.u[i + 8] = hi[i]; }
  return t.v;
}

// B-fragment (16-bit 32x16): per-lane K = kh*16 + 0..15 contiguous.
// Caller passes base + kh*16; 32B-aligned at all call sites.
__device__ __forceinline__ v16bf ldB(const unsigned short* p) {
  BF16x16 t;
  t.u = *(const u16x16*)p;
  return t.v;
}

__device__ __forceinline__ v8f wmma_bf16(v16bf a, v16bf b, v8f c) {
  return __builtin_amdgcn_wmma_f32_16x16x32_bf16(
      false, a, false, b, (short)0, c, false, false);
}

// Async Global -> LDS copy of 16 bytes per lane (ASYNCcnt-tracked).
// lds_off = byte offset within the workgroup LDS allocation (low 32 bits of
// a generic pointer to a __shared__ object, per the LDS aperture rules).
__device__ __forceinline__ void async_ld_b128(unsigned lds_off,
                                              const unsigned short* g) {
  asm volatile("global_load_async_to_lds_b128 %0, %1, off"
               :: "v"(lds_off), "v"((unsigned long long)(size_t)g)
               : "memory");
}

__device__ __forceinline__ unsigned ldsAddr(const void* p) {
  return (unsigned)(unsigned long long)(size_t)p;
}

// ---------------- converters ----------------

__global__ void cvt_f32_bf16(const float* __restrict__ in,
                             unsigned short* __restrict__ out, int n) {
  int i = blockIdx.x * blockDim.x + threadIdx.x;
  if (i < n) out[i] = f2bf(in[i]);
}

// W[K=HID][N=HID] fp32 -> Wt[N][K] bf16 (transpose so B columns are K-contiguous)
__global__ void cvt_w_transpose(const float* __restrict__ W,
                                unsigned short* __restrict__ Wt) {
  int i = blockIdx.x * blockDim.x + threadIdx.x;   // over HID*HID
  int n = i >> 10, k = i & (HID - 1);
  Wt[(size_t)n * HID + k] = f2bf(W[(size_t)k * HID + n]);
}

// ---------------- WMMA GEMM: [MTOT,HID] x Wt[HID,HID]^T + bias ----------------
// One wave computes a 32x64 output tile (2 A-frags x 4 B-frags -> 8 wmma per
// 32-wide K step; 1.5 b128 loads per wmma). mode: 0 = bf16 token-major,
// 1 = bf16 transposed (Vt[n][m], dim-major for P*V), 2 = fp32 token-major.
__global__ __launch_bounds__(128) void gemm_wmma_k(
    const unsigned short* __restrict__ A,
    const unsigned short* __restrict__ Wt,
    const float* __restrict__ bias,
    unsigned short* __restrict__ outb,
    float* __restrict__ outf,
    int mode)
{
  const int wid  = threadIdx.x >> 5;
  const int lane = threadIdx.x & 31;
  const int tile = blockIdx.x * 4 + wid;      // 128 * 16 = 2048 wave tiles
  const int tM   = (tile >> 4) << 5;          // 32-row tile
  const int tN   = (tile & 15) << 6;          // 64-col tile
  const int row  = lane & 15;
  const int kh   = lane >> 4;

  const unsigned short* ar0 = A  + (size_t)(tM +  0 + row) * HID;
  const unsigned short* ar1 = A  + (size_t)(tM + 16 + row) * HID;
  const unsigned short* br0 = Wt + (size_t)(tN +  0 + row) * HID;
  const unsigned short* br1 = Wt + (size_t)(tN + 16 + row) * HID;
  const unsigned short* br2 = Wt + (size_t)(tN + 32 + row) * HID;
  const unsigned short* br3 = Wt + (size_t)(tN + 48 + row) * HID;

  v8f c00 = {}, c01 = {}, c02 = {}, c03 = {};
  v8f c10 = {}, c11 = {}, c12 = {}, c13 = {};

  for (int k0 = 0; k0 < HID; k0 += 32) {
    // prefetch next K chunk of every stream into GL2
    __builtin_prefetch(ar0 + k0 + 256, 0, 3);
    __builtin_prefetch(ar1 + k0 + 256, 0, 3);
    __builtin_prefetch(br0 + k0 + 256, 0, 3);
    __builtin_prefetch(br1 + k0 + 256, 0, 3);
    __builtin_prefetch(br2 + k0 + 256, 0, 3);
    __builtin_prefetch(br3 + k0 + 256, 0, 3);

    v16bf a0 = ldA(ar0 + k0 + kh * 16);
    v16bf a1 = ldA(ar1 + k0 + kh * 16);

    v16bf b0 = ldB(br0 + k0 + kh * 16);
    c00 = wmma_bf16(a0, b0, c00);
    c10 = wmma_bf16(a1, b0, c10);
    v16bf b1 = ldB(br1 + k0 + kh * 16);
    c01 = wmma_bf16(a0, b1, c01);
    c11 = wmma_bf16(a1, b1, c11);
    v16bf b2 = ldB(br2 + k0 + kh * 16);
    c02 = wmma_bf16(a0, b2, c02);
    c12 = wmma_bf16(a1, b2, c12);
    v16bf b3 = ldB(br3 + k0 + kh * 16);
    c03 = wmma_bf16(a0, b3, c03);
    c13 = wmma_bf16(a1, b3, c13);
  }

#pragma unroll
  for (int s = 0; s < 2; ++s) {
    const int mb = tM + s * 16 + kh * 8;      // C/D frag: M = r + 8*kh
#pragma unroll
    for (int j = 0; j < 4; ++j) {
      v8f acc;
      if (s == 0) acc = (j == 0) ? c00 : (j == 1) ? c01 : (j == 2) ? c02 : c03;
      else        acc = (j == 0) ? c10 : (j == 1) ? c11 : (j == 2) ? c12 : c13;
      const int n = tN + j * 16 + row;        // C/D frag: N = lane&15
      const float bn = bias[n];
      if (mode == 0) {
#pragma unroll
        for (int r = 0; r < 8; ++r)
          outb[(size_t)(mb + r) * HID + n] = f2bf(acc[r] + bn);
      } else if (mode == 1) {
        u16x8 pk;
#pragma unroll
        for (int r = 0; r < 8; ++r) pk[r] = f2bf(acc[r] + bn);
        *(u16x8*)(outb + (size_t)n * MTOT + mb) = pk;   // Vt[dim][token]
      } else {
#pragma unroll
        for (int r = 0; r < 8; ++r)
          outf[(size_t)(mb + r) * HID + n] = acc[r] + bn;
      }
    }
  }
}

// ---------------- flash attention ----------------
// Block = 4 waves with identical (b, h) and consecutive 16-query tiles.
// K/V chunks (32 keys) are staged ONCE per block into LDS with async
// global->LDS copies (ASYNCcnt), then all 4 waves read WMMA B-fragments
// from LDS. This removes the 4x redundant global K/V streams.
__global__ __launch_bounds__(128) void attn_wmma_k(
    const unsigned short* __restrict__ Qb,   // [MTOT,HID] bf16
    const unsigned short* __restrict__ Kb,   // [MTOT,HID] bf16
    const unsigned short* __restrict__ Vt,   // [HID,MTOT] bf16 (dim-major)
    unsigned short* __restrict__ ctx)        // [MTOT,HID] bf16
{
  __shared__ __align__(16) unsigned short kbuf[32 * 64];     // K chunk: key-major
  __shared__ __align__(16) unsigned short vbuf[64 * 32];     // V chunk: dim-major
  __shared__ __align__(16) unsigned short pbuf[4 * 16 * 32]; // per-wave P tile

  const int tid  = threadIdx.x;
  const int wid  = tid >> 5;
  const int lane = tid & 31;
  const int task = blockIdx.x * 4 + wid;     // 2*16*128 = 4096 tasks
  const int qt   = task & 127;               // fastest: 4 waves share (b,h)
  const int h    = (task >> 7) & 15;
  const int b    = task >> 11;
  const int row  = lane & 15;
  const int kh   = lane >> 4;

  // cooperative staging indices (whole block)
  const int krow    = tid >> 2;              // 0..31 : key within chunk
  const int kcolseg = (tid & 3) << 4;        // 0/16/32/48 : dim element offset
  const int vrow    = tid >> 1;              // 0..63 : dim within head
  const int vseg    = (tid & 1) << 4;        // 0/16 : key element offset
  const unsigned short* kg = Kb + (size_t)(b * SEQ + krow) * HID + h * HEADD + kcolseg;
  const unsigned short* vg = Vt + (size_t)(h * HEADD + vrow) * MTOT + b * SEQ + vseg;
  const unsigned kl0 = ldsAddr(&kbuf[krow * 64 + kcolseg]);
  const unsigned vl0 = ldsAddr(&vbuf[vrow * 32 + vseg]);

  const unsigned short* qrow = Qb + (size_t)(b * SEQ + qt * 16 + row) * HID + h * HEADD;
  const v16bf aq0 = ldA(qrow +  0 + kh * 16);   // Q tile 16x64, K-dim halves
  const v16bf aq1 = ldA(qrow + 32 + kh * 16);

  float mi[8], li[8];
  v8f o0 = {}, o1 = {}, o2 = {}, o3 = {};
#pragma unroll
  for (int r = 0; r < 8; ++r) { mi[r] = -3.0e38f; li[r] = 0.0f; }

  unsigned short* pw = pbuf + wid * (16 * 32);

  for (int kc = 0; kc < SEQ; kc += 32) {
    // ---- stage K (32x64) and V (64x32) chunks: 16B x2 per lane each ----
    async_ld_b128(kl0,      kg + (size_t)kc * HID);
    async_ld_b128(kl0 + 16, kg + (size_t)kc * HID + 8);
    async_ld_b128(vl0,      vg + kc);
    async_ld_b128(vl0 + 16, vg + kc + 8);
    asm volatile("s_wait_asynccnt 0" ::: "memory");
    __syncthreads();                         // staged data visible to all waves

    // ---- scores: S = Q (16x64) x K^T, two 16-key tiles, from LDS ----
    v8f s0 = {}, s1 = {};
    s0 = wmma_bf16(aq0, ldB(kbuf + (row)      * 64 +  0 + kh * 16), s0);
    s0 = wmma_bf16(aq1, ldB(kbuf + (row)      * 64 + 32 + kh * 16), s0);
    s1 = wmma_bf16(aq0, ldB(kbuf + (16 + row) * 64 +  0 + kh * 16), s1);
    s1 = wmma_bf16(aq1, ldB(kbuf + (16 + row) * 64 + 32 + kh * 16), s1);

    // online softmax over this 32-key chunk; row reductions across the
    // 16-lane half-wave that owns rows (r + 8*kh)
#pragma unroll
    for (int r = 0; r < 8; ++r) {
      float x0 = s0[r] * SCALE, x1 = s1[r] * SCALE;
      float mx = fmaxf(x0, x1);
      mx = fmaxf(mx, __shfl_xor(mx, 8, 32));
      mx = fmaxf(mx, __shfl_xor(mx, 4, 32));
      mx = fmaxf(mx, __shfl_xor(mx, 2, 32));
      mx = fmaxf(mx, __shfl_xor(mx, 1, 32));
      float mn = fmaxf(mi[r], mx);
      float al = __expf(mi[r] - mn);
      float p0 = __expf(x0 - mn);
      float p1 = __expf(x1 - mn);
      float sm = p0 + p1;
      sm += __shfl_xor(sm, 8, 32);
      sm += __shfl_xor(sm, 4, 32);
      sm += __shfl_xor(sm, 2, 32);
      sm += __shfl_xor(sm, 1, 32);
      li[r] = li[r] * al + sm;
      mi[r] = mn;
      o0[r] *= al; o1[r] *= al; o2[r] *= al; o3[r] *= al;
      // stage P (c-frag layout) into LDS: row = r+8*kh, col = key-in-chunk
      pw[(r + 8 * kh) * 32 + row]      = f2bf(p0);
      pw[(r + 8 * kh) * 32 + 16 + row] = f2bf(p1);
    }

    asm volatile("s_wait_dscnt 0" ::: "memory");   // P stores visible wave-wide

    // reload P as an A-fragment (16x32)
    const v16bf ap = ldA(pw + row * 32 + kh * 16);

    // ---- O += P x V, V fragments from LDS (dim-major) ----
    o0 = wmma_bf16(ap, ldB(vbuf + ( 0 + row) * 32 + kh * 16), o0);
    o1 = wmma_bf16(ap, ldB(vbuf + (16 + row) * 32 + kh * 16), o1);
    o2 = wmma_bf16(ap, ldB(vbuf + (32 + row) * 32 + kh * 16), o2);
    o3 = wmma_bf16(ap, ldB(vbuf + (48 + row) * 32 + kh * 16), o3);

    __syncthreads();   // all reads done before next chunk overwrites LDS
  }

  // normalize and write context (token-major bf16)
  unsigned short* cb = ctx + (size_t)(b * SEQ + qt * 16) * HID + h * HEADD;
#pragma unroll
  for (int r = 0; r < 8; ++r) {
    const float inv = 1.0f / li[r];
    const size_t off = (size_t)(8 * kh + r) * HID + row;
    cb[off +  0] = f2bf(o0[r] * inv);
    cb[off + 16] = f2bf(o1[r] * inv);
    cb[off + 32] = f2bf(o2[r] * inv);
    cb[off + 48] = f2bf(o3[r] * inv);
  }
}

// ---------------- host-side orchestration ----------------

extern "C" void kernel_launch(void* const* d_in, const int* in_sizes, int n_in,
                              void* d_out, int out_size, void* d_ws, size_t ws_size,
                              hipStream_t stream) {
  const float* q  = (const float*)d_in[0];
  const float* k  = (const float*)d_in[1];
  const float* v  = (const float*)d_in[2];
  const float* Wq = (const float*)d_in[3];
  const float* bq = (const float*)d_in[4];
  const float* Wk = (const float*)d_in[5];
  const float* bk = (const float*)d_in[6];
  const float* Wv = (const float*)d_in[7];
  const float* bv = (const float*)d_in[8];
  const float* Wo = (const float*)d_in[9];
  const float* bo = (const float*)d_in[10];
  float* out = (float*)d_out;

  const size_t T = (size_t)MTOT * HID;   // 4,194,304 activation elems
  const size_t W = (size_t)HID * HID;    // 1,048,576 weight elems

  unsigned short* ws  = (unsigned short*)d_ws;
  unsigned short* qin = ws;              // bf16 staging
  unsigned short* kin = qin + T;
  unsigned short* vin = kin + T;
  unsigned short* wqT = vin + T;
  unsigned short* wkT = wqT + W;
  unsigned short* wvT = wkT + W;
  unsigned short* woT = wvT + W;
  unsigned short* Qp  = woT + W;
  unsigned short* Kp  = Qp + T;
  unsigned short* Vt  = Kp + T;
  unsigned short* ctx = Vt + T;          // total 64 MiB of ws

  const int cvtBlocks = (int)(T / 256);  // 16384
  cvt_f32_bf16<<<cvtBlocks, 256, 0, stream>>>(q, qin, (int)T);
  cvt_f32_bf16<<<cvtBlocks, 256, 0, stream>>>(k, kin, (int)T);
  cvt_f32_bf16<<<cvtBlocks, 256, 0, stream>>>(v, vin, (int)T);

  const int wBlocks = (int)(W / 256);    // 4096
  cvt_w_transpose<<<wBlocks, 256, 0, stream>>>(Wq, wqT);
  cvt_w_transpose<<<wBlocks, 256, 0, stream>>>(Wk, wkT);
  cvt_w_transpose<<<wBlocks, 256, 0, stream>>>(Wv, wvT);
  cvt_w_transpose<<<wBlocks, 256, 0, stream>>>(Wo, woT);

  // 2048 wave-tiles / 4 waves per block = 512 blocks
  gemm_wmma_k<<<512, 128, 0, stream>>>(qin, wqT, bq, Qp, nullptr, 0);
  gemm_wmma_k<<<512, 128, 0, stream>>>(kin, wkT, bk, Kp, nullptr, 0);
  gemm_wmma_k<<<512, 128, 0, stream>>>(vin, wvT, bv, Vt, nullptr, 1);

  attn_wmma_k<<<1024, 128, 0, stream>>>(Qp, Kp, Vt, ctx);

  gemm_wmma_k<<<512, 128, 0, stream>>>(ctx, woT, bo, nullptr, out, 2);
}
